// GCN_62122406969972
// MI455X (gfx1250) — matrix operand order repaired and checked by reading, not compiled, not run on hardware
//
#include <hip/hip_runtime.h>
#include <hip/hip_bf16.h>

#define N_NODES 100000
#define HID 128
#define N_EDGES 1600000
#define BN_EPS 1e-5f

typedef __attribute__((ext_vector_type(16))) _Float16 v16h;
typedef __attribute__((ext_vector_type(8)))  _Float16 v8h;
typedef __attribute__((ext_vector_type(8)))  float    v8f;

// ---------------------------------------------------------------------------
// degree: in-degree count via f32 atomics (self-loop added in rsqrt kernel)
// ---------------------------------------------------------------------------
__global__ __launch_bounds__(256) void gcn_degree_kernel(
    const long long* __restrict__ dst, float* __restrict__ deg, int E) {
  int i = blockIdx.x * 256 + threadIdx.x;
  if (i < E) unsafeAtomicAdd(&deg[(int)dst[i]], 1.0f);
}

__global__ __launch_bounds__(256) void gcn_dinv_kernel(
    const float* __restrict__ deg, float* __restrict__ dinv, int n) {
  int i = blockIdx.x * 256 + threadIdx.x;
  if (i < n) dinv[i] = rsqrtf(deg[i] + 1.0f);  // +1 = self loop
}

// ---------------------------------------------------------------------------
// h = x @ W  via v_wmma_f32_16x16x32_f16 (f16 inputs, f32 accumulate)
// block = 256 thr (8 waves). Block owns 32 rows (2 M-tiles per wave);
// wave w owns cols [16w, 16w+16). W^T staged once per block in LDS as f16 so
// B fragments are contiguous ds_load_b128s, each reused by 2 WMMAs.
// 100000 rows = 3125 blocks * 32, exact.
// ---------------------------------------------------------------------------
__global__ __launch_bounds__(256) void gcn_gemm_wmma_kernel(
    const float* __restrict__ X, const float* __restrict__ W,
    float* __restrict__ H) {
  __shared__ _Float16 wt[HID * HID];  // wt[n*128 + k] = W[k][n]
  const int tid = threadIdx.x;
  for (int idx = tid; idx < HID * HID; idx += 256) {
    int k = idx >> 7, n = idx & 127;
    wt[n * HID + k] = (_Float16)W[idx];
  }
  __syncthreads();

  const int wave  = tid >> 5;        // col tile 0..7
  const int lane  = tid & 31;
  const int lhalf = lane >> 4;       // 0 | 1
  const int l16   = lane & 15;

  const int row0 = blockIdx.x * 32;                 // block's first row
  const int ncol = wave * 16 + l16;                 // B/D column this lane feeds
  const float*    xrow0 = X + (size_t)(row0 + l16) * HID;       // M-tile 0
  const float*    xrow1 = X + (size_t)(row0 + 16 + l16) * HID;  // M-tile 1
  const _Float16* wrow  = wt + ncol * HID;

  v8f acc0 = {}, acc1 = {};
#pragma unroll
  for (int kk = 0; kk < 4; ++kk) {
    // B fragment (32x16): lane n holds K = kbB..kbB+7, kbB+16..kbB+23, kbB = kk*32 + 8*(lane>=16)
    const int kbB = kk * 32 + lhalf * 8;
    v8h b0 = *(const v8h*)(wrow + kbB);
    v8h b1 = *(const v8h*)(wrow + kbB + 16);
    v16h b;
#pragma unroll
    for (int e = 0; e < 8; ++e) { b[e] = b0[e]; b[e + 8] = b1[e]; }

    // A fragments (16-bit A 16x32 layout): elems 0..7 -> K = kb..kb+7,
    // elems 8..15 -> K = kb+16..kb+23, kb = kk*32 + 8*(lane>=16)
    const int kb = kk * 32 + lhalf * 8;
    v16h a0, a1;
#pragma unroll
    for (int e = 0; e < 8; ++e) {
      a0[e]     = (_Float16)xrow0[kb + e];
      a0[e + 8] = (_Float16)xrow0[kb + 16 + e];
      a1[e]     = (_Float16)xrow1[kb + e];
      a1[e + 8] = (_Float16)xrow1[kb + 16 + e];
    }

    acc0 = __builtin_amdgcn_wmma_f32_16x16x32_f16(
        false, a0, false, b, (short)0, acc0, false, false);
    acc1 = __builtin_amdgcn_wmma_f32_16x16x32_f16(
        false, a1, false, b, (short)0, acc1, false, false);
  }

  // D layout: VGPR r -> row = rowtile + r + 8*(lane>=16), col = ncol
  float* hb0 = H + ((size_t)(row0 + lhalf * 8)) * HID + ncol;
  float* hb1 = H + ((size_t)(row0 + 16 + lhalf * 8)) * HID + ncol;
#pragma unroll
  for (int r = 0; r < 8; ++r) {
    hb0[(size_t)r * HID] = acc0[r];
    hb1[(size_t)r * HID] = acc1[r];
  }
}

// ---------------------------------------------------------------------------
// scatter: one wave per edge, lane handles 4 channels (float4),
// agg[dst] += h[src] * dinv[src]*dinv[dst]   (L2-resident atomics)
// ---------------------------------------------------------------------------
__global__ __launch_bounds__(256) void gcn_scatter_kernel(
    const long long* __restrict__ src, const long long* __restrict__ dst,
    const float* __restrict__ dinv, const float* __restrict__ H,
    float* __restrict__ AGG, int E) {
  const int eid = blockIdx.x * 8 + (threadIdx.x >> 5);
  if (eid >= E) return;
  const int s = (int)src[eid];
  const int d = (int)dst[eid];
  const float norm = dinv[s] * dinv[d];
  const int c = (threadIdx.x & 31) * 4;
  const float4 v = *(const float4*)(H + (size_t)s * HID + c);
  float* out = AGG + (size_t)d * HID + c;
  unsafeAtomicAdd(out + 0, v.x * norm);
  unsafeAtomicAdd(out + 1, v.y * norm);
  unsafeAtomicAdd(out + 2, v.z * norm);
  unsafeAtomicAdd(out + 3, v.w * norm);
}

// ---------------------------------------------------------------------------
// self-loop + bias + BatchNorm(eval) + ReLU + optional residual
// ---------------------------------------------------------------------------
__global__ __launch_bounds__(256) void gcn_bn_relu_kernel(
    const float* __restrict__ AGG, const float* __restrict__ H,
    const float* __restrict__ dinv, const float* __restrict__ bias,
    const float* __restrict__ gamma, const float* __restrict__ beta,
    const float* __restrict__ mean, const float* __restrict__ var,
    const float* __restrict__ xprev, float* __restrict__ out, int total) {
  int i = blockIdx.x * 256 + threadIdx.x;
  if (i >= total) return;
  const int c = i & 127;
  const int n = i >> 7;
  const float di  = dinv[n];
  const float val = AGG[i] + H[i] * di * di + bias[c];
  const float xb  = (val - mean[c]) * rsqrtf(var[c] + BN_EPS) * gamma[c] + beta[c];
  float r = fmaxf(xb, 0.0f);
  if (xprev) r += xprev[i];
  out[i] = r;
}

// ---------------------------------------------------------------------------
extern "C" void kernel_launch(void* const* d_in, const int* in_sizes, int n_in,
                              void* d_out, int out_size, void* d_ws, size_t ws_size,
                              hipStream_t stream) {
  const long long* edge = (const long long*)d_in[0];  // [2, E] int64
  const float* emb    = (const float*)d_in[1];        // [N, 128]
  const float* Ws     = (const float*)d_in[2];        // [3, 128, 128]
  const float* bs     = (const float*)d_in[3];        // [3, 128]
  const float* gammas = (const float*)d_in[4];
  const float* betas  = (const float*)d_in[5];
  const float* rmeans = (const float*)d_in[6];
  const float* rvars  = (const float*)d_in[7];
  float* out = (float*)d_out;

  const long long* e_src = edge;
  const long long* e_dst = edge + N_EDGES;

  // workspace carve-up
  char* ws = (char*)d_ws;
  float* deg  = (float*)ws;                 ws += (size_t)N_NODES * 4;
  float* dinv = (float*)ws;                 ws += (size_t)N_NODES * 4;
  float* h    = (float*)ws;                 ws += (size_t)N_NODES * HID * 4;
  float* agg  = (float*)ws;                 ws += (size_t)N_NODES * HID * 4;
  float* bufA = (float*)ws;                 ws += (size_t)N_NODES * HID * 4;
  float* bufB = (float*)ws;                 ws += (size_t)N_NODES * HID * 4;

  const int total = N_NODES * HID;
  const int gemm_blocks    = N_NODES / 32;           // 3125, exact
  const int scatter_blocks = (N_EDGES + 7) / 8;      // wave per edge
  const int ew_blocks      = (total + 255) / 256;
  const int deg_blocks     = (N_EDGES + 255) / 256;
  const int n_blocks       = (N_NODES + 255) / 256;

  // degrees + dinv (recomputed every call: deterministic, no cached state)
  hipMemsetAsync(deg, 0, (size_t)N_NODES * 4, stream);
  gcn_degree_kernel<<<deg_blocks, 256, 0, stream>>>(e_dst, deg, N_EDGES);
  gcn_dinv_kernel<<<n_blocks, 256, 0, stream>>>(deg, dinv, N_NODES);

  const float* x_in[3]   = { emb,  bufA, bufB };
  const float* x_res[3]  = { nullptr, bufA, bufB };
  float*       x_out[3]  = { bufA, bufB, out };

  for (int L = 0; L < 3; ++L) {
    gcn_gemm_wmma_kernel<<<gemm_blocks, 256, 0, stream>>>(
        x_in[L], Ws + (size_t)L * HID * HID, h);
    hipMemsetAsync(agg, 0, (size_t)total * 4, stream);
    gcn_scatter_kernel<<<scatter_blocks, 256, 0, stream>>>(
        e_src, e_dst, dinv, h, agg, N_EDGES);
    gcn_bn_relu_kernel<<<ew_blocks, 256, 0, stream>>>(
        agg, h, dinv, bs + L * HID, gammas + L * HID, betas + L * HID,
        rmeans + L * HID, rvars + L * HID, x_res[L], x_out[L], total);
  }
}
